// MSA_15358803051064
// MI455X (gfx1250) — matrix operand order repaired and checked by reading, not compiled, not run on hardware
//
#include <hip/hip_runtime.h>
#include <hip/hip_bf16.h>
#include <math.h>

// ---- problem constants -----------------------------------------------------
static constexpr int Bn   = 4;
static constexpr int L1   = 2048;
static constexpr int L2   = 2048;
static constexpr int NHID = 1024;   // = K = N for all GEMMs
static constexpr int NH   = 16;
static constexpr int HD   = 64;

typedef _Float16 half8  __attribute__((ext_vector_type(8)));
typedef _Float16 half16 __attribute__((ext_vector_type(16)));
typedef float    float8 __attribute__((ext_vector_type(8)));

__device__ __forceinline__ half16 combine8(const _Float16* lo, const _Float16* hi) {
    half8 a = *(const half8*)lo;
    half8 b = *(const half8*)hi;
    half16 r;
#pragma unroll
    for (int i = 0; i < 8; ++i) { r[i] = a[i]; r[i + 8] = b[i]; }
    return r;
}

// ---- 16-lane butterfly reductions -----------------------------------------
// Rows of the WMMA C layout live across 16-lane groups; reduce with
// v_permlane16_b32 (pure VALU) instead of ds_bpermute to keep the DS pipe
// free for matrix fragment traffic.
#if defined(__has_builtin) && __has_builtin(__builtin_amdgcn_permlane16)
__device__ __forceinline__ float lanesel16(float x, unsigned s0, unsigned s1) {
    unsigned u = __float_as_uint(x);
    u = __builtin_amdgcn_permlane16(u, u, s0, s1, false, false);
    return __uint_as_float(u);
}
__device__ __forceinline__ float red_max16(float x) {
    x = fmaxf(x, lanesel16(x, 0x67452301u, 0xEFCDAB89u));   // xor 1
    x = fmaxf(x, lanesel16(x, 0x54761032u, 0xDCFE98BAu));   // xor 2
    x = fmaxf(x, lanesel16(x, 0x32107654u, 0xBA98FEDCu));   // xor 4
    x = fmaxf(x, lanesel16(x, 0xFEDCBA98u, 0x76543210u));   // xor 8
    return x;
}
__device__ __forceinline__ float red_sum16(float x) {
    x += lanesel16(x, 0x67452301u, 0xEFCDAB89u);
    x += lanesel16(x, 0x54761032u, 0xDCFE98BAu);
    x += lanesel16(x, 0x32107654u, 0xBA98FEDCu);
    x += lanesel16(x, 0xFEDCBA98u, 0x76543210u);
    return x;
}
#else
__device__ __forceinline__ float red_max16(float x) {
#pragma unroll
    for (int off = 1; off < 16; off <<= 1) x = fmaxf(x, __shfl_xor(x, off, 32));
    return x;
}
__device__ __forceinline__ float red_sum16(float x) {
#pragma unroll
    for (int off = 1; off < 16; off <<= 1) x += __shfl_xor(x, off, 32);
    return x;
}
#endif

// ---- stage 1: fp32 -> f16 converts ----------------------------------------
__global__ void cvt_f16_kernel(const float* __restrict__ src, _Float16* __restrict__ dst, int n) {
    int i = blockIdx.x * blockDim.x + threadIdx.x;
    if (i < n) dst[i] = (_Float16)src[i];
}

// dst[n][k] = src[k][n], 1024x1024
__global__ void cvt_t_f16_kernel(const float* __restrict__ src, _Float16* __restrict__ dst) {
    int i = blockIdx.x * blockDim.x + threadIdx.x;      // i = n*1024 + k
    int n = i >> 10, k = i & 1023;
    dst[i] = (_Float16)src[k * NHID + n];
}

// ---- stage 2/4: WMMA GEMM, M=2048 K=1024 N=1024 per batch ------------------
// MODE 0: Q  -> outH [B][NH][L][HD], val=(acc+bias)*scale
// MODE 1: K  -> outH [B][NH][L][HD]
// MODE 2: V  -> outH [B][NH][HD][L]   (transposed for PV B-fragments)
// MODE 3: O  -> outF [B][L][NHID] fp32, val=acc+bias
template <int MODE>
__global__ __launch_bounds__(256)
void gemm_wmma_kernel(const _Float16* __restrict__ Aall,   // [B][2048][1024]
                      const _Float16* __restrict__ Wt,     // [1024 n][1024 k] (W transposed)
                      const float*    __restrict__ bias,   // [1024]
                      _Float16* __restrict__ outH,
                      float*    __restrict__ outF,
                      float scale) {
    constexpr int Kd = NHID, M = 2048, Lq = 2048;
    const int b  = blockIdx.z;
    const int bm = blockIdx.y * 64;
    const int bn = blockIdx.x * 256;
    const _Float16* A = Aall + (size_t)b * M * Kd;

    // double-buffered tiles: one barrier per K-step
    __shared__ __align__(64) _Float16 Asm[2][64][32];
    __shared__ __align__(64) _Float16 Bsm[2][256][32];

    const int tid  = threadIdx.x;
    const int lane = tid & 31;
    const int w    = tid >> 5;
    const int wm   = w & 1;        // 2 waves along M
    const int wn   = w >> 1;       // 4 waves along N
    const int mrow = lane & 15;
    const int kg   = lane >> 4;
    const int arow = tid >> 2, aseg = tid & 3;

    float8 acc[2][4];
#pragma unroll
    for (int mi = 0; mi < 2; ++mi)
#pragma unroll
        for (int ni = 0; ni < 4; ++ni) acc[mi][ni] = (float8){0,0,0,0,0,0,0,0};

    half8 aR;
    half8 bR[4];
    auto gload = [&](int kk) {
        aR = *(const half8*)&A[(size_t)(bm + arow) * Kd + kk + aseg * 8];
#pragma unroll
        for (int i = 0; i < 4; ++i) {
            int c = tid + 256 * i;
            int col = c >> 2, seg = c & 3;
            bR[i] = *(const half8*)&Wt[(size_t)(bn + col) * Kd + kk + seg * 8];
        }
    };
    auto lstore = [&](int buf) {
        *(half8*)&Asm[buf][arow][aseg * 8] = aR;
#pragma unroll
        for (int i = 0; i < 4; ++i) {
            int c = tid + 256 * i;
            int col = c >> 2, seg = c & 3;
            *(half8*)&Bsm[buf][col][seg * 8] = bR[i];
        }
    };

    gload(0);
    lstore(0);
    __syncthreads();

    constexpr int STEPS = Kd / 32;
#pragma unroll 1
    for (int s = 0; s < STEPS; ++s) {
        const int cur = s & 1;
        if (s + 1 < STEPS) gload((s + 1) * 32);   // overlap next fill with compute

        half16 af[2];
#pragma unroll
        for (int mi = 0; mi < 2; ++mi) {
            const _Float16* rp = &Asm[cur][wm * 32 + mi * 16 + mrow][0];
            af[mi] = combine8(rp + kg * 8, rp + 16 + kg * 8);
        }
#pragma unroll
        for (int ni = 0; ni < 4; ++ni) {
            const _Float16* cp = &Bsm[cur][wn * 64 + ni * 16 + mrow][0];
            half16 bf = combine8(cp + kg * 8, cp + 16 + kg * 8);
#pragma unroll
            for (int mi = 0; mi < 2; ++mi)
                acc[mi][ni] = __builtin_amdgcn_wmma_f32_16x16x32_f16(
                    false, af[mi], false, bf, (short)0, acc[mi][ni], false, false);
        }

        if (s + 1 < STEPS) {
            lstore((s + 1) & 1);
            __syncthreads();
        }
    }

    // epilogue (MODE is compile-time)
    const int hi = lane >> 4;
    const int nl = lane & 15;
#pragma unroll
    for (int mi = 0; mi < 2; ++mi) {
#pragma unroll
        for (int ni = 0; ni < 4; ++ni) {
            int col = bn + wn * 64 + ni * 16 + nl;
            float bc = bias[col];
            int h  = col >> 6;      // col / 64
            int hd = col & 63;
#pragma unroll
            for (int r = 0; r < 8; ++r) {
                int row = bm + wm * 32 + mi * 16 + r + 8 * hi;
                float v = acc[mi][ni][r];
                if (MODE == 0) {
                    outH[(((size_t)b * NH + h) * Lq + row) * HD + hd] =
                        (_Float16)((v + bc) * scale);
                } else if (MODE == 1) {
                    outH[(((size_t)b * NH + h) * Lq + row) * HD + hd] = (_Float16)(v + bc);
                } else if (MODE == 2) {
                    outH[(((size_t)b * NH + h) * HD + hd) * Lq + row] = (_Float16)(v + bc);
                } else {
                    outF[((size_t)b * Lq + row) * NHID + col] = v + bc;
                }
            }
        }
    }
}

// ---- stage 3: flash attention ---------------------------------------------
// grid: (L1/128, NH, B), block 256 (8 waves), each wave owns 16 q rows.
// Q was pre-scaled by log2(e)/sqrt(HD), so scores are in the log2 domain and
// every softmax exponential is a single v_exp_f32 (exp2), no mul by log2e.
// 64-key tiles amortize the row reductions; no block-level barriers (the P
// bounce is per-wave and the LDS pipeline is in-order per wave).
__global__ __launch_bounds__(256)
void attn_wmma_kernel(const _Float16* __restrict__ qh,   // [B][NH][L1][HD]
                      const _Float16* __restrict__ kh,   // [B][NH][L2][HD]
                      const _Float16* __restrict__ vh,   // [B][NH][HD][L2]
                      const unsigned char* __restrict__ mask,  // [B][L2]
                      _Float16* __restrict__ attn) {     // [B][L1][NH*HD]
    const int b = blockIdx.z, h = blockIdx.y;
    const int tid = threadIdx.x, lane = tid & 31, w = tid >> 5;
    const int qb = blockIdx.x * 128 + w * 16;
    const int mrow = lane & 15, kg = lane >> 4, hi = kg, nl = lane & 15;

    const _Float16* qbase = qh + ((size_t)(b * NH + h)) * L1 * HD;
    const _Float16* kbase = kh + ((size_t)(b * NH + h)) * L2 * HD;
    const _Float16* vbase = vh + ((size_t)(b * NH + h)) * HD * L2;

    __shared__ __align__(64) _Float16 Pl[8][16][64];

    // Q fragments (K=64 -> two 16x32 A-frags), kept in registers
    half16 aq[2];
#pragma unroll
    for (int c = 0; c < 2; ++c) {
        const _Float16* rp = qbase + (size_t)(qb + mrow) * HD + c * 32;
        aq[c] = combine8(rp + kg * 8, rp + 16 + kg * 8);
    }

    float8 o[4];
#pragma unroll
    for (int ni = 0; ni < 4; ++ni) o[ni] = (float8){0,0,0,0,0,0,0,0};
    float rm[8], rl[8];
#pragma unroll
    for (int r = 0; r < 8; ++r) { rm[r] = -3.0e38f; rl[r] = 0.0f; }

    const float8 zeroc = (float8){0,0,0,0,0,0,0,0};

#pragma unroll 1
    for (int kb = 0; kb < L2; kb += 64) {
        // prefetch next key/value tiles into cache (global_prefetch_b8)
        if (kb + 64 < L2) {
            __builtin_prefetch(kbase + (size_t)(kb + 64 + lane) * HD, 0, 0);
            __builtin_prefetch(kbase + (size_t)(kb + 96 + lane) * HD, 0, 0);
            __builtin_prefetch(vbase + (size_t)lane * L2 + kb + 64, 0, 0);
            __builtin_prefetch(vbase + (size_t)(32 + lane) * L2 + kb + 64, 0, 0);
        }

        // ---- S = Q @ K^T for 64 keys (four 16x16 tiles) -------------------
        // batch all K-fragment loads, then the WMMA burst
        half16 bk[4][2];
        unsigned char mk[4];
#pragma unroll
        for (int t = 0; t < 4; ++t) {
            int key = kb + t * 16 + nl;
            const _Float16* krp = kbase + (size_t)key * HD;
            bk[t][0] = combine8(krp + kg * 8, krp + 16 + kg * 8);            // K-dim 0..31
            bk[t][1] = combine8(krp + 32 + kg * 8, krp + 48 + kg * 8);  // 32..63
            mk[t] = mask[(size_t)b * L2 + key];
        }
        float8 s[4];
#pragma unroll
        for (int t = 0; t < 4; ++t) {
            s[t] = __builtin_amdgcn_wmma_f32_16x16x32_f16(false, aq[0], false, bk[t][0],
                                                          (short)0, zeroc, false, false);
            s[t] = __builtin_amdgcn_wmma_f32_16x16x32_f16(false, aq[1], false, bk[t][1],
                                                          (short)0, s[t], false, false);
#pragma unroll
            for (int r = 0; r < 8; ++r) s[t][r] = mk[t] ? -1.0e8f : s[t][r];
        }

        // ---- online softmax row update (log2 domain) ----------------------
        float corr[8];
#pragma unroll
        for (int r = 0; r < 8; ++r) {
            float mloc = fmaxf(fmaxf(s[0][r], s[1][r]), fmaxf(s[2][r], s[3][r]));
            mloc = red_max16(mloc);
            float mnew = fmaxf(rm[r], mloc);
            corr[r] = exp2f(rm[r] - mnew);
            float ls = 0.0f;
#pragma unroll
            for (int t = 0; t < 4; ++t) {
                float p = exp2f(s[t][r] - mnew);
                s[t][r] = p;
                ls += p;
            }
            ls = red_sum16(ls);
            rl[r] = rl[r] * corr[r] + ls;
            rm[r] = mnew;
        }
#pragma unroll
        for (int ni = 0; ni < 4; ++ni)
#pragma unroll
            for (int r = 0; r < 8; ++r) o[ni][r] *= corr[r];

        // ---- P: C-layout -> A-fragment layout via per-wave LDS bounce -----
        __builtin_amdgcn_wave_barrier();
#pragma unroll
        for (int t = 0; t < 4; ++t)
#pragma unroll
            for (int r = 0; r < 8; ++r)
                Pl[w][r + 8 * hi][t * 16 + nl] = (_Float16)s[t][r];
        __builtin_amdgcn_wave_barrier();
        half16 pf[2];
#pragma unroll
        for (int c = 0; c < 2; ++c) {
            const _Float16* pp = &Pl[w][mrow][c * 32];
            pf[c] = combine8(pp + kg * 8, pp + 16 + kg * 8);
        }
        __builtin_amdgcn_wave_barrier();

        // ---- O += P @ V  (V transposed: rows = hd, contiguous over keys) --
#pragma unroll
        for (int ni = 0; ni < 4; ++ni) {
            const _Float16* vrp = vbase + (size_t)(ni * 16 + nl) * L2 + kb;
#pragma unroll
            for (int c = 0; c < 2; ++c) {
                half16 bv = combine8(vrp + c * 32 + kg * 8, vrp + c * 32 + 16 + kg * 8);
                o[ni] = __builtin_amdgcn_wmma_f32_16x16x32_f16(false, pf[c], false, bv,
                                                               (short)0, o[ni], false, false);
            }
        }
    }

    float inv[8];
#pragma unroll
    for (int r = 0; r < 8; ++r) inv[r] = 1.0f / rl[r];
#pragma unroll
    for (int ni = 0; ni < 4; ++ni)
#pragma unroll
        for (int r = 0; r < 8; ++r) {
            int row = qb + r + 8 * hi;
            attn[((size_t)b * L1 + row) * NHID + h * HD + ni * 16 + nl] =
                (_Float16)(o[ni][r] * inv[r]);
        }
}

// ---- host side -------------------------------------------------------------
extern "C" void kernel_launch(void* const* d_in, const int* in_sizes, int n_in,
                              void* d_out, int out_size, void* d_ws, size_t ws_size,
                              hipStream_t stream) {
    (void)in_sizes; (void)n_in; (void)out_size; (void)ws_size;
    const float* inp1 = (const float*)d_in[0];
    const float* inp2 = (const float*)d_in[1];
    const unsigned char* mask = (const unsigned char*)d_in[2];
    const float* Wq = (const float*)d_in[3];
    const float* bq = (const float*)d_in[4];
    const float* Wk = (const float*)d_in[5];
    const float* bk = (const float*)d_in[6];
    const float* Wv = (const float*)d_in[7];
    const float* bv = (const float*)d_in[8];
    const float* Wo = (const float*)d_in[9];
    const float* bo = (const float*)d_in[10];
    float* out = (float*)d_out;

    const size_t ACT = (size_t)Bn * L1 * NHID;   // 8388608
    const size_t WSZ = (size_t)NHID * NHID;      // 1048576
    _Float16* base   = (_Float16*)d_ws;
    _Float16* inp1h  = base;
    _Float16* inp2h  = inp1h + ACT;
    _Float16* WqT    = inp2h + ACT;
    _Float16* WkT    = WqT + WSZ;
    _Float16* WvT    = WkT + WSZ;
    _Float16* WoT    = WvT + WSZ;
    _Float16* qhB    = WoT + WSZ;
    _Float16* khB    = qhB + ACT;
    _Float16* vhB    = khB + ACT;
    _Float16* attnh  = vhB + ACT;

    // stage 1: converts
    cvt_f16_kernel<<<(int)(ACT / 256), 256, 0, stream>>>(inp1, inp1h, (int)ACT);
    cvt_f16_kernel<<<(int)(ACT / 256), 256, 0, stream>>>(inp2, inp2h, (int)ACT);
    cvt_t_f16_kernel<<<(int)(WSZ / 256), 256, 0, stream>>>(Wq, WqT);
    cvt_t_f16_kernel<<<(int)(WSZ / 256), 256, 0, stream>>>(Wk, WkT);
    cvt_t_f16_kernel<<<(int)(WSZ / 256), 256, 0, stream>>>(Wv, WvT);
    cvt_t_f16_kernel<<<(int)(WSZ / 256), 256, 0, stream>>>(Wo, WoT);

    dim3 ggrid(NHID / 256, 2048 / 64, Bn);   // (4, 32, 4)
    // stage 2: projections
    // Q is scaled by log2(e)/sqrt(HD) so attention exponentials are pure exp2.
    const float qscale = 0.125f * 1.44269504088896340736f;
    gemm_wmma_kernel<0><<<ggrid, 256, 0, stream>>>(inp1h, WqT, bq, qhB, nullptr, qscale);
    gemm_wmma_kernel<1><<<ggrid, 256, 0, stream>>>(inp2h, WkT, bk, khB, nullptr, 1.0f);
    gemm_wmma_kernel<2><<<ggrid, 256, 0, stream>>>(inp2h, WvT, bv, vhB, nullptr, 1.0f);

    // stage 3: attention
    dim3 agrid(L1 / 128, NH, Bn);            // (16, 16, 4)
    attn_wmma_kernel<<<agrid, 256, 0, stream>>>(qhB, khB, vhB, mask, attnh);

    // stage 4: output projection -> fp32 d_out
    gemm_wmma_kernel<3><<<ggrid, 256, 0, stream>>>(attnh, WoT, bo, nullptr, out, 1.0f);
}
